// Bahdanau_Attention_89824946029023
// MI455X (gfx1250) — matrix operand order repaired and checked by reading, not compiled, not run on hardware
//
#include <hip/hip_runtime.h>
#include <hip/hip_bf16.h>

// Bahdanau attention, B=64 S=2048 DEC_H=1024 ENC2=1024.
// Dominant op: 131072x1024x1024 GEMM (274 GFLOP) -> bf16 WMMA with f32 accum.
// Encoder (512 MB, streamed once per pass) uses non-temporal loads so the
// reused We_bf16 (2 MB) stays resident in WGP$/L2.

#define B_DIM 64
#define S_DIM 2048
#define DH    1024        // DEC_H == N dim of the big GEMM
#define E2    1024        // ENC2  == K dim of the big GEMM
#define WROW  (DH + E2)   // W_w row length (2048)
#define NEG_BIG (-1.0e10f)

typedef __attribute__((ext_vector_type(16))) __bf16 v16bf;
typedef __attribute__((ext_vector_type(8)))  float  v8f;
typedef __attribute__((ext_vector_type(4)))  float  f32x4;

__device__ __forceinline__ unsigned short f2bf(float f) {
  unsigned u = __float_as_uint(f);
  u += 0x7FFFu + ((u >> 16) & 1u);        // round-to-nearest-even
  return (unsigned short)(u >> 16);
}

// ---------------------------------------------------------------------------
// bias[b][d] = dec_hidden[b] . W_w[d, 0:1024] + W_b[d]
// grid 256 (4 blocks per b), block 256
// ---------------------------------------------------------------------------
__global__ void k_prep_bias(const float* __restrict__ dec_hidden,
                            const float* __restrict__ W_w,
                            const float* __restrict__ W_b,
                            float* __restrict__ bias) {
  __shared__ float sh[DH];
  const int b    = blockIdx.x >> 2;
  const int dblk = blockIdx.x & 3;
  for (int i = threadIdx.x; i < DH; i += 256) sh[i] = dec_hidden[b * DH + i];
  __syncthreads();
  const int d = dblk * 256 + threadIdx.x;
  const float* wr = W_w + (size_t)d * WROW;
  float acc = W_b[d];
  for (int k = 0; k < DH; k += 4) {
    acc += sh[k] * wr[k] + sh[k + 1] * wr[k + 1] +
           sh[k + 2] * wr[k + 2] + sh[k + 3] * wr[k + 3];
  }
  bias[b * DH + d] = acc;
}

// ---------------------------------------------------------------------------
// We_bf16[n][k] = bf16(W_w[n][1024 + k]); row-major 1024x1024 (2 MB, L2-resident)
// grid 1024, block 256, 4 elements/thread
// ---------------------------------------------------------------------------
__global__ void k_conv_we(const float* __restrict__ W_w,
                          unsigned short* __restrict__ Web) {
  const int idx = blockIdx.x * 256 + threadIdx.x;   // 0 .. 262143
  const int n  = idx >> 8;                          // 256 float4 groups per row
  const int kg = idx & 255;
  const f32x4 f = *(const f32x4*)(W_w + (size_t)n * WROW + DH + kg * 4);
  const unsigned lo = f2bf(f.x) | ((unsigned)f2bf(f.y) << 16);
  const unsigned hi = f2bf(f.z) | ((unsigned)f2bf(f.w) << 16);
  *(uint2*)(Web + (size_t)n * E2 + kg * 4) = make_uint2(lo, hi);
}

// ---------------------------------------------------------------------------
// Scores: per (b, 128-row s-tile) workgroup.
//   enc tile (128x1024 f32) -> bf16 in LDS once (non-temporal reads), then
//   D = A(16x32) x B(32x16) WMMA over K=1024, N=1024 (8 passes x 8 tiles/wave),
//   fused tanh(acc + bias) . v  -> masked raw scores.
// LDS: 128 x 1032 bf16 = 258 KB (pad keeps ds_load_b128 16B-aligned,
// row bank step = 4 -> conflict-free across the 16 row-lanes).
// Inner loop: all 8 B fragments loaded first (16 global_load_b128 in flight),
// then 8 back-to-back WMMAs -> deep load pipeline despite 2 waves/SIMD.
// ---------------------------------------------------------------------------
__global__ __launch_bounds__(256)
void k_scores(const float* __restrict__ enc,
              const unsigned short* __restrict__ Web,
              const float* __restrict__ bias,
              const float* __restrict__ v_w,
              const int*   __restrict__ mask,
              float* __restrict__ scores) {
  extern __shared__ unsigned short smemA[];
  const int LDA   = 1032;
  const int b     = blockIdx.y;
  const int s0    = blockIdx.x * 128;
  const int tid   = threadIdx.x;

  // ---- stage A tile: f32 -> bf16 into LDS (streaming, non-temporal) ----
  {
    const float* src = enc + ((size_t)b * S_DIM + s0) * (size_t)E2;
    for (int it = tid; it < 128 * 256; it += 256) {
      const int row = it >> 8;
      const int cg  = it & 255;
      const f32x4 f =
          __builtin_nontemporal_load((const f32x4*)(src + (size_t)row * E2 + cg * 4));
      const unsigned lo = f2bf(f.x) | ((unsigned)f2bf(f.y) << 16);
      const unsigned hi = f2bf(f.z) | ((unsigned)f2bf(f.w) << 16);
      *(uint2*)(smemA + row * LDA + cg * 4) = make_uint2(lo, hi);
    }
  }
  __syncthreads();

  const int lane = tid & 31;
  const int wave = tid >> 5;      // 0..7, owns s-rows [wave*16, wave*16+16)
  const int h    = lane >> 4;     // lane half
  const int r    = lane & 15;
  const int m0   = wave * 16;

  float sp[8];
#pragma unroll
  for (int j = 0; j < 8; ++j) sp[j] = 0.0f;

  const unsigned short* aRow = smemA + (m0 + r) * LDA;  // A: lane r = row

  for (int pass = 0; pass < 8; ++pass) {                // N in chunks of 128
    v8f acc[8];
    const v8f zero = {0.f, 0.f, 0.f, 0.f, 0.f, 0.f, 0.f, 0.f};
#pragma unroll
    for (int nt = 0; nt < 8; ++nt) acc[nt] = zero;

    // B row base for this pass: lane column n = pass*128 + nt*16 + r,
    // half h covers K = kk + h*8 .. +7 and kk + 16 + h*8 .. +7
    const unsigned short* bBase = Web + (size_t)(pass * 128 + r) * E2 + h * 8;

    for (int ks = 0; ks < 32; ++ks) {                   // K in chunks of 32
      const int kk = ks * 32;
      // A fragment (16-bit A 16x32 layout): elems {h*8..h*8+7, 16+h*8..+7}
      union Frag { uint4 u[2]; v16bf v; };
      Frag A;
      A.u[0] = *(const uint4*)(aRow + kk + h * 8);
      A.u[1] = *(const uint4*)(aRow + kk + 16 + h * 8);

      // load ALL B fragments first -> 16 b128 loads in flight
      Frag Bf[8];
#pragma unroll
      for (int nt = 0; nt < 8; ++nt) {
        const unsigned short* bp = bBase + (size_t)(nt * 16) * E2 + kk;
        Bf[nt].u[0] = *(const uint4*)(bp);
        Bf[nt].u[1] = *(const uint4*)(bp + 16);
      }
      // then drain with back-to-back WMMAs
#pragma unroll
      for (int nt = 0; nt < 8; ++nt) {
        acc[nt] = __builtin_amdgcn_wmma_f32_16x16x32_bf16(
            false, A.v, false, Bf[nt].v, (short)0, acc[nt], false, false);
      }
    }

    // fused epilogue: sp[m] += v[n] * tanh(acc[m][n] + bias[b][n])
#pragma unroll
    for (int nt = 0; nt < 8; ++nt) {
      const int n  = pass * 128 + nt * 16 + r;
      const float bn = bias[b * DH + n];
      const float vn = v_w[n];
#pragma unroll
      for (int j = 0; j < 8; ++j) {         // acc row m = j + 8*h
        sp[j] += vn * tanhf(acc[nt][j] + bn);
      }
    }
  }

  // reduce over the 16 column-lanes within each half
#pragma unroll
  for (int j = 0; j < 8; ++j) {
    float v = sp[j];
    v += __shfl_xor(v, 1, 32);
    v += __shfl_xor(v, 2, 32);
    v += __shfl_xor(v, 4, 32);
    v += __shfl_xor(v, 8, 32);
    sp[j] = v;
  }
  if (r == 0) {                             // lanes 0 and 16 of each wave
#pragma unroll
    for (int j = 0; j < 8; ++j) {
      const int s = s0 + m0 + j + 8 * h;
      float sc = sp[j];
      if (mask[b * S_DIM + s] == 0) sc = NEG_BIG;
      scores[b * S_DIM + s] = sc;
    }
  }
}

// ---------------------------------------------------------------------------
// Softmax over S per batch row, normalized in place. grid 64, block 256.
// ---------------------------------------------------------------------------
__global__ void k_softmax(float* __restrict__ scores) {
  __shared__ float red[256];
  const int b   = blockIdx.x;
  const int tid = threadIdx.x;
  float loc[8];
  float mx = -3.4e38f;
#pragma unroll
  for (int i = 0; i < 8; ++i) {
    loc[i] = scores[b * S_DIM + tid + 256 * i];
    mx = fmaxf(mx, loc[i]);
  }
  red[tid] = mx; __syncthreads();
  for (int off = 128; off > 0; off >>= 1) {
    if (tid < off) red[tid] = fmaxf(red[tid], red[tid + off]);
    __syncthreads();
  }
  mx = red[0]; __syncthreads();
  float sum = 0.f;
#pragma unroll
  for (int i = 0; i < 8; ++i) { loc[i] = __expf(loc[i] - mx); sum += loc[i]; }
  red[tid] = sum; __syncthreads();
  for (int off = 128; off > 0; off >>= 1) {
    if (tid < off) red[tid] += red[tid + off];
    __syncthreads();
  }
  const float inv = 1.0f / red[0];
#pragma unroll
  for (int i = 0; i < 8; ++i) scores[b * S_DIM + tid + 256 * i] = loc[i] * inv;
}

// ---------------------------------------------------------------------------
// Context partials: part[sc][b][e] = sum_{s in chunk sc} w[b][s]*enc[b][s][e]
// grid (8, 64), block 256; coalesced streaming pass over encoder (512 MB, NT).
// ---------------------------------------------------------------------------
__global__ void k_context(const float* __restrict__ enc,
                          const float* __restrict__ w,
                          float* __restrict__ part) {
  __shared__ float ws[256];
  const int b   = blockIdx.y;
  const int sc  = blockIdx.x;
  const int tid = threadIdx.x;
  ws[tid] = w[b * S_DIM + sc * 256 + tid];
  __syncthreads();
  float a0 = 0.f, a1 = 0.f, a2 = 0.f, a3 = 0.f;
  const float* base = enc + ((size_t)b * S_DIM + sc * 256) * (size_t)E2;
  for (int s = 0; s < 256; ++s) {
    const float wv = ws[s];
    const float* row = base + (size_t)s * E2;
    a0 += wv * __builtin_nontemporal_load(row + tid);
    a1 += wv * __builtin_nontemporal_load(row + tid + 256);
    a2 += wv * __builtin_nontemporal_load(row + tid + 512);
    a3 += wv * __builtin_nontemporal_load(row + tid + 768);
  }
  float* p = part + ((size_t)sc * B_DIM + b) * E2;
  p[tid]       = a0;
  p[tid + 256] = a1;
  p[tid + 512] = a2;
  p[tid + 768] = a3;
}

// Deterministic reduction of the 8 s-chunk partials. grid 256, block 256.
__global__ void k_ctx_reduce(const float* __restrict__ part,
                             float* __restrict__ ctx) {
  const int idx = blockIdx.x * 256 + threadIdx.x;   // 0 .. 65535
  float s = 0.f;
#pragma unroll
  for (int sc = 0; sc < 8; ++sc) s += part[(size_t)sc * B_DIM * E2 + idx];
  ctx[idx] = s;
}

// ---------------------------------------------------------------------------
extern "C" void kernel_launch(void* const* d_in, const int* in_sizes, int n_in,
                              void* d_out, int out_size, void* d_ws, size_t ws_size,
                              hipStream_t stream) {
  const float* dec_hidden = (const float*)d_in[0];   // (1,64,1024)
  const float* enc        = (const float*)d_in[1];   // (64,2048,1024)
  const int*   mask       = (const int*)  d_in[2];   // (64,1,2048)
  const float* W_w        = (const float*)d_in[3];   // (1024,2048)
  const float* W_b        = (const float*)d_in[4];   // (1024,)
  const float* v_w        = (const float*)d_in[5];   // (1,1024)

  float* ctx   = (float*)d_out;                       // context: 64*1024
  float* atten = (float*)d_out + B_DIM * E2;          // weights: 64*2048 (scores in place)

  unsigned short* Web = (unsigned short*)d_ws;                          // 2 MB
  float* bias = (float*)((char*)d_ws + (size_t)DH * E2 * 2);            // 256 KB
  float* part = (float*)((char*)d_ws + (size_t)DH * E2 * 2
                                     + (size_t)B_DIM * DH * 4);         // 2 MB

  k_prep_bias<<<256, 256, 0, stream>>>(dec_hidden, W_w, W_b, bias);
  k_conv_we <<<1024, 256, 0, stream>>>(W_w, Web);

  const size_t shmem = 128 * 1032 * sizeof(unsigned short);  // 258 KB < 320 KB
  k_scores  <<<dim3(16, 64), 256, shmem, stream>>>(enc, Web, bias, v_w, mask, atten);

  k_softmax <<<64, 256, 0, stream>>>(atten);
  k_context <<<dim3(8, 64), 256, 0, stream>>>(enc, atten, part);
  k_ctx_reduce<<<256, 256, 0, stream>>>(part, ctx);
}
